// SqueezedLiGRU_S_43585328120311
// MI455X (gfx1250) — matrix operand order repaired and checked by reading, not compiled, not run on hardware
//
#include <hip/hip_runtime.h>
#include <math.h>
#include <stdint.h>

// ---------------------------------------------------------------------------
// SqueezedLiGRU on MI455X (gfx1250, wave32, WMMA bf16 16x16x32, f32 accum)
//
//   w_ln = LN( x @ M ),  M = blockdiag(w_in) @ w_rec_W^T      (fused 1+2+3)
//   h    = ligru_scan(w_ln, u_rec)   (single persistent WGP, LDS-resident h)
//   out  = grouped_linear(h, w_out)  (per-group WMMA, K=64)
//
// Round-4 change: tensor_load_to_lds uses the 6-arg builtin signature
// (this toolchain's clang expects (g0, g1, g2, g3, g4, cpol)).
// ---------------------------------------------------------------------------

#define B_   32
#define T_   1000
#define D_   512
#define H_   512
#define G_   8
#define H2_  1024
#define BT_  (B_ * T_)

typedef __attribute__((ext_vector_type(16))) __bf16 v16bf;
typedef __attribute__((ext_vector_type(8)))  __bf16 v8bf;
typedef __attribute__((ext_vector_type(8)))  float  v8f;
typedef __attribute__((ext_vector_type(4)))  unsigned int u32x4;
typedef __attribute__((ext_vector_type(8)))  int i32x8;
typedef __attribute__((ext_vector_type(4)))  int i32x4;

__device__ __forceinline__ unsigned short f2bf(float f) {
    union { float f; unsigned int u; } c; c.f = f;
    unsigned int u = c.u;
    u += 0x7FFFu + ((u >> 16) & 1u);     // round-to-nearest-even
    return (unsigned short)(u >> 16);
}

// A-fragment (16x32 bf16, ISA 7.12.2 layout) from a row-major bf16 row.
// lanes 0-15 : row = lane,     K = {0..7, 16..23}  of the 32-wide k-tile
// lanes 16-31: row = lane-16,  K = {8..15, 24..31}
__device__ __forceinline__ v16bf load_afrag(const unsigned short* rowptr, int kbase) {
    v8bf lo = *(const v8bf*)(const void*)(rowptr + kbase);
    v8bf hi = *(const v8bf*)(const void*)(rowptr + kbase + 16);
    v16bf a;
#pragma unroll
    for (int i = 0; i < 8; ++i) { a[i] = lo[i]; a[8 + i] = hi[i]; }
    return a;
}

__device__ __forceinline__ v8f wmma_bf16(v16bf a, v16bf b, v8f c) {
    // (neg_a, A, neg_b, B, c_mod, C, reuse_a, reuse_b)
    return __builtin_amdgcn_wmma_f32_16x16x32_bf16(false, a, false, b,
                                                   (short)0, c, false, false);
}

// ---------------------------------------------------------------------------
// K1a: M[d][j] = sum_ho w_in[g, d%64, ho] * w_rec_W[j, g*64+ho],  g = d/64
//      packed directly into the 32-lane B-fragment layout:
//      block (jt*16 + ktile) holds 32 lanes x 16 contiguous bf16 (32B/lane).
// ---------------------------------------------------------------------------
__global__ void __launch_bounds__(256) k_pack_M(const float* __restrict__ w_in,
                                                const float* __restrict__ w_rec,
                                                unsigned short* __restrict__ Mpack) {
    int idx = blockIdx.x * 256 + threadIdx.x;       // 0 .. 512*1024-1
    int d = idx & 511;                              // K index (input feature)
    int j = idx >> 9;                               // N index (gate col, 0..1023)
    int g = d >> 6, i = d & 63;
    const float* wi = w_in + (size_t)(g * 64 + i) * 64;
    const float* wr = w_rec + (size_t)j * 512 + g * 64;
    float s = 0.f;
#pragma unroll 8
    for (int ho = 0; ho < 64; ++ho) s += wi[ho] * wr[ho];
    int ktile = d >> 5, krem = d & 31;
    int lane = ((krem >> 3) & 1) * 16 + (j & 15);
    int blk = (j >> 4) * 16 + ktile;
    Mpack[((size_t)blk * 32 + lane) * 16 + (krem & 7) + ((krem >> 4) << 3)] = f2bf(s);
}

// K1b: uT[k][j] = u_rec[j][k], same B-fragment packing (K=512, N=1024)
__global__ void __launch_bounds__(256) k_pack_uT(const float* __restrict__ u_rec,
                                                 unsigned short* __restrict__ uTp) {
    int idx = blockIdx.x * 256 + threadIdx.x;       // 0 .. 1024*512-1
    int k = idx & 511;
    int j = idx >> 9;
    float v = u_rec[(size_t)j * 512 + k];
    int ktile = k >> 5, krem = k & 31;
    int lane = ((krem >> 3) & 1) * 16 + (j & 15);
    int blk = (j >> 4) * 16 + ktile;
    uTp[((size_t)blk * 32 + lane) * 16 + (krem & 7) + ((krem >> 4) << 3)] = f2bf(v);
}

// K1c: w_out[g][i][o] packed per-group (K=i, N=o), block = (g*4+jt)*2+ktile
__global__ void __launch_bounds__(256) k_pack_wout(const float* __restrict__ w_out,
                                                   unsigned short* __restrict__ wpk) {
    int idx = blockIdx.x * 256 + threadIdx.x;       // 0 .. 8*64*64-1
    int g = idx >> 12;
    int i = (idx >> 6) & 63;
    int o = idx & 63;
    float v = w_out[idx];
    int ktile = i >> 5, krem = i & 31;
    int lane = ((krem >> 3) & 1) * 16 + (o & 15);
    int blk = (g * 4 + (o >> 4)) * 2 + ktile;
    wpk[((size_t)blk * 32 + lane) * 16 + (krem & 7) + ((krem >> 4) << 3)] = f2bf(v);
}

// K1d: x (f32, row-major [BT,512]) -> bf16
__global__ void __launch_bounds__(256) k_conv_x(const float* __restrict__ x,
                                                unsigned short* __restrict__ xb) {
    size_t idx = (size_t)blockIdx.x * 256 + threadIdx.x;
    xb[idx] = f2bf(x[idx]);
}

// ---------------------------------------------------------------------------
// K2: w_ln = LayerNorm( x @ M )    16 rows x 1024 cols per block, K = 512
//     8 waves; wave w owns col-tiles jt = w*8 .. w*8+7  (128 cols)
//     A tile (16x512 bf16 = 16 KB, contiguous) staged into LDS by the TDM.
// ---------------------------------------------------------------------------
__global__ void __launch_bounds__(256) k_gemm_ln(const unsigned short* __restrict__ xb,
                                                 const unsigned short* __restrict__ Mpack,
                                                 const float* __restrict__ gamma,
                                                 const float* __restrict__ beta,
                                                 float* __restrict__ wln) {
    const int lane = threadIdx.x & 31;
    const int wv   = threadIdx.x >> 5;        // 0..7
    const int r0   = blockIdx.x * 16;         // row base in [0, BT)

    __shared__ unsigned short s_atile[16 * 512];   // 16 KB bf16 A tile
    __shared__ float s_sum[16];
    __shared__ float s_ss[16];
    if (threadIdx.x < 16) { s_sum[threadIdx.x] = 0.f; s_ss[threadIdx.x] = 0.f; }

#if __has_builtin(__builtin_amdgcn_tensor_load_to_lds) && \
    __has_builtin(__builtin_amdgcn_s_wait_tensorcnt)
    // --- Tensor Data Mover: one DMA of the whole 16-row x 1024 B tile ------
    // D# group0: count=1 | lds_addr | global_addr[56:0] | type=2 ("image")
    // D# group1: data_size=8B; tensor_dim0 = tile_dim0 = stride0 = 128 units;
    //            tensor_dim1 = tile_dim1 = 16 rows; groups 2/3/4 unused (2-D).
    if (threadIdx.x < 32) {                        // wave 0 only (EXEC ignored)
        unsigned long long ga =
            (unsigned long long)(uintptr_t)(const void*)(xb + (size_t)r0 * 512);
        unsigned lds_addr = (unsigned)(uintptr_t)(void*)&s_atile[0];
        u32x4 g0;
        g0[0] = 1u;                                // count=1, user mode
        g0[1] = lds_addr;                          // LDS byte address
        g0[2] = (unsigned)ga;                      // global_addr[31:0]
        g0[3] = (unsigned)(ga >> 32) | (2u << 30); // global_addr[56:32] | type=2
        i32x8 g1;
        g1[0] = 0x00030000;                        // data_size=3 (8 bytes)
        g1[1] = (int)(128u << 16);                 // tensor_dim0[15:0] = 128
        g1[2] = (int)(16u << 16);                  // tensor_dim0 hi=0 | tensor_dim1 lo=16
        g1[3] = (int)(128u << 16);                 // tensor_dim1 hi=0 | tile_dim0=128
        g1[4] = 16;                                // tile_dim1=16 | tile_dim2=0
        g1[5] = 128;                               // tensor_dim0_stride = 128 units
        g1[6] = 0;
        g1[7] = 0;
        i32x4 gz4 = {0, 0, 0, 0};
        i32x8 gz8 = {0, 0, 0, 0, 0, 0, 0, 0};
        __builtin_amdgcn_tensor_load_to_lds(g0, g1, gz4, gz4, gz8, 0);
        __builtin_amdgcn_s_wait_tensorcnt(0);
    }
#else
    // Fallback: cooperative staged copy (32 bf16 per thread)
    {
        const unsigned short* src = xb + (size_t)r0 * 512 + threadIdx.x * 32;
        unsigned short* dst = &s_atile[threadIdx.x * 32];
#pragma unroll
        for (int i = 0; i < 4; ++i)
            *(v8bf*)(void*)(dst + i * 8) = *(const v8bf*)(const void*)(src + i * 8);
    }
#endif
    __syncthreads();

    v8f acc[8];
#pragma unroll
    for (int q = 0; q < 8; ++q)
#pragma unroll
        for (int e = 0; e < 8; ++e) acc[q][e] = 0.f;

    const unsigned short* arow = &s_atile[(size_t)(lane & 15) * 512];
    const int khalf = (lane >> 4) * 8;
    // B base for (q=0, kt=0): block jt0*16 = wv*8*16; +q -> +8192 elems, +kt -> +512 elems
    const unsigned short* bbase = Mpack + ((size_t)(wv * 8) * 16 * 32 + lane) * 16;

    for (int kt = 0; kt < 16; ++kt) {
        v16bf bfr[8];
#pragma unroll
        for (int q = 0; q < 8; ++q)
            bfr[q] = *(const v16bf*)(const void*)(bbase + (size_t)q * 8192 + (size_t)kt * 512);
        v16bf afrag = load_afrag(arow, kt * 32 + khalf);
#pragma unroll
        for (int q = 0; q < 8; ++q)
            acc[q] = wmma_bf16(afrag, bfr[q], acc[q]);
    }

    // LayerNorm over the full 1024-wide row: C-fragment row = v + 8*(lane>>4),
    // col = jt*16 + (lane&15). Reduce 16 col-lanes via shfl_xor, then one
    // LDS float atomic per row per wave, then barrier.
    const int rhalf = (lane >> 4) * 8;
#pragma unroll
    for (int v = 0; v < 8; ++v) {
        float s = 0.f, ss = 0.f;
#pragma unroll
        for (int q = 0; q < 8; ++q) { float xv = acc[q][v]; s += xv; ss += xv * xv; }
#pragma unroll
        for (int m = 1; m < 16; m <<= 1) {
            s  += __shfl_xor(s,  m, 32);
            ss += __shfl_xor(ss, m, 32);
        }
        if ((lane & 15) == 0) {
            atomicAdd(&s_sum[rhalf + v], s);
            atomicAdd(&s_ss[rhalf + v], ss);
        }
    }
    __syncthreads();

#pragma unroll
    for (int v = 0; v < 8; ++v) {
        int row = rhalf + v;
        float mu   = s_sum[row] * (1.f / (float)H2_);
        float var  = s_ss[row] * (1.f / (float)H2_) - mu * mu;
        float rstd = rsqrtf(var + 1e-5f);
#pragma unroll
        for (int q = 0; q < 8; ++q) {
            int col = (wv * 8 + q) * 16 + (lane & 15);
            float val = (acc[q][v] - mu) * rstd * gamma[col] + beta[col];
            wln[(size_t)(r0 + row) * H2_ + col] = val;
        }
    }
}

// ---------------------------------------------------------------------------
// K3: LiGRU scan. One persistent workgroup (512 threads = 16 waves) owns
// h[32x512]: bf16 copy in LDS (A-operand), f32 master in registers.
// Wave p (0..15) owns gate-column tiles {p, p+16} (a) / {p+32, p+48} (z)
// for BOTH 16-row tiles (B-fragments reused across row tiles).
// Per step: 16 k-tiles x 8 WMMAs per wave = 2048 WMMAs on the WGP.
// ---------------------------------------------------------------------------
__global__ void __launch_bounds__(512) k_scan(const float* __restrict__ wln,
                                              const unsigned short* __restrict__ uTp,
                                              unsigned short* __restrict__ h_all) {
    __shared__ unsigned short h_bf[32 * 512];     // 32 KB bf16 h state

    const int tid  = threadIdx.x;
    const int lane = tid & 31;
    const int p    = tid >> 5;                    // wave id = column index

#pragma unroll
    for (int i = 0; i < 32; ++i) h_bf[tid * 32 + i] = 0;   // h0 = 0
    __syncthreads();

    float hreg[2][2][8];                          // [rowtile][a-coltile][slot]
#pragma unroll
    for (int rt = 0; rt < 2; ++rt)
#pragma unroll
        for (int q = 0; q < 2; ++q)
#pragma unroll
            for (int v = 0; v < 8; ++v) hreg[rt][q][v] = 0.f;

    const int khalf = (lane >> 4) * 8;
    const int rhalf = (lane >> 4) * 8;
    const unsigned short* hrow0 = &h_bf[(size_t)(lane & 15) * 512];
    const unsigned short* hrow1 = &h_bf[(size_t)(16 + (lane & 15)) * 512];

    // Hoisted B-fragment base pointers: block (jt*16 + kt), +kt = +512 elems,
    // jt in {p, p+16, p+32, p+48} -> deltas of 16*16*512 = 131072 elems.
    const unsigned short* bpA0 = uTp + ((size_t)p * 16 * 32 + lane) * 16;
    const unsigned short* bpA1 = bpA0 + (size_t)16 * 16 * 512;
    const unsigned short* bpZ0 = bpA0 + (size_t)32 * 16 * 512;
    const unsigned short* bpZ1 = bpA0 + (size_t)48 * 16 * 512;

    // coalesced writeback mapping: thread owns h_bf[tid*32 .. tid*32+31]
    const int wb_b   = (tid * 32) >> 9;           // batch row
    const int wb_col = (tid * 32) & 511;

    for (int t = 0; t < T_; ++t) {
        v8f accA[2][2], accZ[2][2];
#pragma unroll
        for (int rt = 0; rt < 2; ++rt)
#pragma unroll
            for (int q = 0; q < 2; ++q)
#pragma unroll
                for (int e = 0; e < 8; ++e) { accA[rt][q][e] = 0.f; accZ[rt][q][e] = 0.f; }

        for (int kt = 0; kt < 16; ++kt) {
            const size_t ko = (size_t)kt * 512;
            v16bf bA0 = *(const v16bf*)(const void*)(bpA0 + ko);
            v16bf bA1 = *(const v16bf*)(const void*)(bpA1 + ko);
            v16bf bZ0 = *(const v16bf*)(const void*)(bpZ0 + ko);
            v16bf bZ1 = *(const v16bf*)(const void*)(bpZ1 + ko);
            v16bf a0 = load_afrag(hrow0, kt * 32 + khalf);
            v16bf a1 = load_afrag(hrow1, kt * 32 + khalf);
            accA[0][0] = wmma_bf16(a0, bA0, accA[0][0]);
            accA[1][0] = wmma_bf16(a1, bA0, accA[1][0]);
            accA[0][1] = wmma_bf16(a0, bA1, accA[0][1]);
            accA[1][1] = wmma_bf16(a1, bA1, accA[1][1]);
            accZ[0][0] = wmma_bf16(a0, bZ0, accZ[0][0]);
            accZ[1][0] = wmma_bf16(a1, bZ0, accZ[1][0]);
            accZ[0][1] = wmma_bf16(a0, bZ1, accZ[0][1]);
            accZ[1][1] = wmma_bf16(a1, bZ1, accZ[1][1]);
        }
        __syncthreads();   // all h_bf reads for step t done

#pragma unroll
        for (int rt = 0; rt < 2; ++rt) {
#pragma unroll
            for (int q = 0; q < 2; ++q) {
#pragma unroll
                for (int v = 0; v < 8; ++v) {
                    int row = rt * 16 + rhalf + v;                 // batch b
                    int col = (p + q * 16) * 16 + (lane & 15);     // 0..511
                    size_t wbase = ((size_t)row * T_ + t) * H2_;
                    float ag = accA[rt][q][v] + wln[wbase + col];
                    float zg = accZ[rt][q][v] + wln[wbase + 512 + col];
                    float z  = 1.f / (1.f + __expf(-zg));
                    float hc = ag > 0.f ? ag : 0.f;
                    float h  = z * hreg[rt][q][v] + (1.f - z) * hc;
                    hreg[rt][q][v] = h;
                    h_bf[(size_t)row * 512 + col] = f2bf(h);
                    if (t + 1 < T_) {
                        __builtin_prefetch(&wln[wbase + H2_ + col], 0, 0);
                        __builtin_prefetch(&wln[wbase + H2_ + 512 + col], 0, 0);
                    }
                }
            }
        }
        __syncthreads();   // h_bf(t+1) visible to all waves

        // coalesced h_all writeback (LDS reads race-free vs. next GEMM reads)
        {
            const unsigned short* src = &h_bf[tid * 32];
            unsigned short* dst = h_all + ((size_t)wb_b * T_ + t) * 512 + wb_col;
#pragma unroll
            for (int i = 0; i < 4; ++i)
                *(v8bf*)(void*)(dst + i * 8) = *(const v8bf*)(const void*)(src + i * 8);
        }
    }
}

// ---------------------------------------------------------------------------
// K4: out = grouped_linear(h, w_out). 16 rows/block, wave g handles group g:
//     4 col tiles x 2 k-tiles = 8 WMMAs per wave.
// ---------------------------------------------------------------------------
__global__ void __launch_bounds__(256) k_out(const unsigned short* __restrict__ h_all,
                                             const unsigned short* __restrict__ wpk,
                                             float* __restrict__ out) {
    const int lane = threadIdx.x & 31;
    const int g    = threadIdx.x >> 5;        // 0..7
    const int r0   = blockIdx.x * 16;

    const unsigned short* arow = h_all + (size_t)(r0 + (lane & 15)) * 512 + g * 64;
    const int khalf = (lane >> 4) * 8;
    const unsigned short* bbase = wpk + (size_t)(g * 4) * 2 * 512 + (size_t)lane * 16;

    v8f acc[4];
#pragma unroll
    for (int jt = 0; jt < 4; ++jt)
#pragma unroll
        for (int e = 0; e < 8; ++e) acc[jt][e] = 0.f;

#pragma unroll
    for (int kt = 0; kt < 2; ++kt) {
        v16bf bfr[4];
#pragma unroll
        for (int jt = 0; jt < 4; ++jt)
            bfr[jt] = *(const v16bf*)(const void*)(bbase + (size_t)(jt * 2 + kt) * 512);
        v16bf afrag = load_afrag(arow, kt * 32 + khalf);
#pragma unroll
        for (int jt = 0; jt < 4; ++jt)
            acc[jt] = wmma_bf16(afrag, bfr[jt], acc[jt]);
    }

    const int rhalf = (lane >> 4) * 8;
#pragma unroll
    for (int jt = 0; jt < 4; ++jt)
#pragma unroll
        for (int v = 0; v < 8; ++v) {
            int row = r0 + rhalf + v;
            int col = g * 64 + jt * 16 + (lane & 15);
            out[(size_t)row * D_ + col] = acc[jt][v];
        }
}

// ---------------------------------------------------------------------------
extern "C" void kernel_launch(void* const* d_in, const int* in_sizes, int n_in,
                              void* d_out, int out_size, void* d_ws, size_t ws_size,
                              hipStream_t stream) {
    (void)in_sizes; (void)n_in; (void)out_size; (void)ws_size;
    const float* x     = (const float*)d_in[0];
    const float* w_in  = (const float*)d_in[1];
    const float* w_rec = (const float*)d_in[2];
    const float* u_rec = (const float*)d_in[3];
    const float* gamma = (const float*)d_in[4];
    const float* beta  = (const float*)d_in[5];
    const float* w_out = (const float*)d_in[6];
    float* out = (float*)d_out;

    char* ws = (char*)d_ws;
    size_t off = 0;
    auto alloc = [&](size_t bytes) -> void* {
        void* p = ws + off;
        off += (bytes + 255) & ~(size_t)255;
        return p;
    };
    unsigned short* xb    = (unsigned short*)alloc((size_t)BT_ * 512 * 2);   // 32.8 MB
    unsigned short* Mpack = (unsigned short*)alloc((size_t)512 * 1024 * 2);  //  1.0 MB
    unsigned short* uTp   = (unsigned short*)alloc((size_t)512 * 1024 * 2);  //  1.0 MB
    unsigned short* wpk   = (unsigned short*)alloc((size_t)8 * 64 * 64 * 2); // 64 KB
    float*          wln   = (float*)alloc((size_t)BT_ * 1024 * 4);           // 131 MB
    unsigned short* hall  = (unsigned short*)alloc((size_t)BT_ * 512 * 2);   // 32.8 MB

    k_pack_M   <<<(512 * 1024) / 256, 256, 0, stream>>>(w_in, w_rec, Mpack);
    k_pack_uT  <<<(1024 * 512) / 256, 256, 0, stream>>>(u_rec, uTp);
    k_pack_wout<<<(8 * 64 * 64) / 256, 256, 0, stream>>>(w_out, wpk);
    k_conv_x   <<<((size_t)BT_ * 512) / 256, 256, 0, stream>>>(x, xb);
    k_gemm_ln  <<<BT_ / 16, 256, 0, stream>>>(xb, Mpack, gamma, beta, wln);
    k_scan     <<<1, 512, 0, stream>>>(wln, uTp, hall);
    k_out      <<<BT_ / 16, 256, 0, stream>>>(hall, wpk, out);
}
